// GNNBlock_73117523247641
// MI455X (gfx1250) — compile-verified, matrix operand order checked by reading, not run on hardware
//
#include <hip/hip_runtime.h>
#include <hip/hip_bf16.h>

typedef __bf16 bf16;
typedef __attribute__((ext_vector_type(16))) __bf16 v16bf;
typedef __attribute__((ext_vector_type(8)))  __bf16 v8bf;
typedef __attribute__((ext_vector_type(8)))  float  v8f;

namespace {
constexpr int   NN   = 100000;    // nodes (3125 blocks of 32 rows)
constexpr int   EE   = 262144;    // edges
constexpr int   DD   = 300;       // emb dim
constexpr int   LL   = 5;         // layers
constexpr int   Dp   = 320;       // padded K for GEMM1 / padded Ncols for GEMM2
constexpr int   HH   = 600;       // hidden
constexpr int   Hp   = 640;       // padded Ncols GEMM1 / padded K GEMM2
constexpr int   CHUNKS = 80;      // BN reduction row chunks (100000/80 = 1250)
constexpr float BN_EPS = 1e-5f;

__host__ __device__ inline size_t cdiv(size_t a, size_t b) { return (a + b - 1) / b; }
}

// ---------------- edge message + scatter: aggr[dst] += relu(x[src] + ea) -----
__global__ void scatter_kernel(const float* __restrict__ x,
                               const float* __restrict__ ea,
                               const int*   __restrict__ ei,
                               float* __restrict__ aggr) {
  size_t idx = (size_t)blockIdx.x * blockDim.x + threadIdx.x;   // E * 75 float4 chunks
  if (idx >= (size_t)EE * (DD / 4)) return;
  int e = (int)(idx / (DD / 4));
  int c = (int)(idx % (DD / 4));
  int src = ei[e];
  int dst = ei[EE + e];
  const float4 xv = *(const float4*)(x  + (size_t)src * DD + c * 4);
  const float4 ev = *(const float4*)(ea + (size_t)e   * DD + c * 4);
  float4 m;
  m.x = fmaxf(xv.x + ev.x, 0.f);
  m.y = fmaxf(xv.y + ev.y, 0.f);
  m.z = fmaxf(xv.z + ev.z, 0.f);
  m.w = fmaxf(xv.w + ev.w, 0.f);
  float* a = aggr + (size_t)dst * DD + c * 4;
  atomicAdd(a + 0, m.x);
  atomicAdd(a + 1, m.y);
  atomicAdd(a + 2, m.z);
  atomicAdd(a + 3, m.w);
}

// ---------------- h0 = (1+eps)*x + aggr, cast to bf16, K-pad to Dp ----------
__global__ void combine_kernel(const float* __restrict__ x,
                               const float* __restrict__ aggr,
                               const float* __restrict__ epsArr, int layer,
                               bf16* __restrict__ xb) {
  size_t idx = (size_t)blockIdx.x * blockDim.x + threadIdx.x;   // N * Dp
  if (idx >= (size_t)NN * Dp) return;
  int row = (int)(idx / Dp);
  int d   = (int)(idx % Dp);
  float v = 0.f;
  if (d < DD) {
    float s = 1.0f + epsArr[layer];
    v = s * x[(size_t)row * DD + d] + aggr[(size_t)row * DD + d];
  }
  xb[idx] = (bf16)v;
}

// ---------------- stage W [K][Nn] (row-major) -> Wt bf16 [Np][Kp] (n-major) --
__global__ void stage_wt(const float* __restrict__ W, bf16* __restrict__ Wt,
                         int K, int Nn, int Kp, int Np) {
  size_t idx = (size_t)blockIdx.x * blockDim.x + threadIdx.x;
  if (idx >= (size_t)Np * Kp) return;
  int n = (int)(idx / Kp);
  int k = (int)(idx % Kp);
  float v = (n < Nn && k < K) ? W[(size_t)k * Nn + n] : 0.f;
  Wt[idx] = (bf16)v;
}

// ---------------- bf16 WMMA GEMM: C[M,Np] = A[M,Kp] * Bt^T + bias -----------
// A: [M][Kp] bf16 row-major.  Bt: [Np][Kp] bf16 (B transposed, n-major).
// One wave per block; wave computes a 32x64 output block:
// 2 M-tiles x 4 N-tiles = 8 WMMA accumulators, A fragments reused across N,
// B fragments reused across M  ->  1.5 b128 loads per v_wmma.
__global__ __launch_bounds__(32)
void gemm_wmma(const bf16* __restrict__ A, const bf16* __restrict__ Bt,
               const float* __restrict__ bias, int biasN,
               float* __restrict__ C, int Kp, int Np) {
  const int lane  = threadIdx.x & 31;
  const int half  = lane >> 4;          // 0: K {0-7,16-23}; 1: K {8-15,24-31}
  const int r     = lane & 15;
  const int mbase = blockIdx.x * 32;    // 2 m-tiles of 16
  const int nbase = blockIdx.y * 64;    // 4 n-tiles of 16

  const bf16* arow0 = A  + (size_t)(mbase + r) * Kp + half * 8;
  const bf16* arow1 = arow0 + (size_t)16 * Kp;
  const bf16* bcol0 = Bt + (size_t)(nbase + r) * Kp + half * 8;

  v8f acc[2][4];
#pragma unroll
  for (int m = 0; m < 2; ++m)
#pragma unroll
    for (int t = 0; t < 4; ++t) acc[m][t] = (v8f){};

  for (int k = 0; k < Kp; k += 32) {
    __builtin_prefetch(arow0 + k + 128, 0, 1);
    __builtin_prefetch(arow1 + k + 128, 0, 1);

    v16bf a[2], b[4];
#pragma unroll
    for (int m = 0; m < 2; ++m) {
      const bf16* p = (m == 0 ? arow0 : arow1) + k;
      v8bf lo = *(const v8bf*)(p);
      v8bf hi = *(const v8bf*)(p + 16);
#pragma unroll
      for (int e = 0; e < 8; ++e) { a[m][e] = lo[e]; a[m][e + 8] = hi[e]; }
    }
#pragma unroll
    for (int t = 0; t < 4; ++t) {
      const bf16* p = bcol0 + (size_t)t * 16 * Kp + k;
      v8bf lo = *(const v8bf*)(p);
      v8bf hi = *(const v8bf*)(p + 16);
#pragma unroll
      for (int e = 0; e < 8; ++e) { b[t][e] = lo[e]; b[t][e + 8] = hi[e]; }
    }
#pragma unroll
    for (int m = 0; m < 2; ++m)
#pragma unroll
      for (int t = 0; t < 4; ++t)
        acc[m][t] = __builtin_amdgcn_wmma_f32_16x16x32_bf16(
            false, a[m], false, b[t], (short)0, acc[m][t], false, false);
  }

#pragma unroll
  for (int t = 0; t < 4; ++t) {
    const int col = nbase + t * 16 + r;
    const float bv = (col < biasN) ? bias[col] : 0.0f;
#pragma unroll
    for (int m = 0; m < 2; ++m) {
#pragma unroll
      for (int e = 0; e < 8; ++e) {
        int row = mbase + m * 16 + half * 8 + e;   // lanes 16-31 hold M+8
        C[(size_t)row * Np + col] = acc[m][t][e] + bv;
      }
    }
  }
}

// ---------------- BN stats, stage 1: per-chunk column partial sums ----------
__global__ void bn_partial(const float* __restrict__ H, int ld, int rowsPerChunk,
                           float* __restrict__ psum, float* __restrict__ psq, int Np) {
  const int c  = threadIdx.x & 15;
  const int rg = threadIdx.x >> 4;           // 16 row-lanes
  const int col = blockIdx.x * 16 + c;
  const int r0  = blockIdx.y * rowsPerChunk;
  float s = 0.f, q = 0.f;
  for (int row = r0 + rg; row < r0 + rowsPerChunk; row += 16) {
    float v = H[(size_t)row * ld + col];
    s += v; q += v * v;
  }
  __shared__ float ss[256], sq[256];
  ss[threadIdx.x] = s; sq[threadIdx.x] = q;
  __syncthreads();
  for (int off = 128; off >= 16; off >>= 1) {
    if (threadIdx.x < off) {
      ss[threadIdx.x] += ss[threadIdx.x + off];
      sq[threadIdx.x] += sq[threadIdx.x + off];
    }
    __syncthreads();
  }
  if (rg == 0) {
    psum[(size_t)blockIdx.y * Np + col] = ss[c];
    psq [(size_t)blockIdx.y * Np + col] = sq[c];
  }
}

// ---------------- BN stats, stage 2: mean + rsqrt(var+eps) ------------------
__global__ void bn_final(const float* __restrict__ psum, const float* __restrict__ psq,
                         int Np, int chunks, float* __restrict__ mean, float* __restrict__ inv) {
  int col = blockIdx.x * blockDim.x + threadIdx.x;
  if (col >= Np) return;
  float s = 0.f, q = 0.f;
  for (int ch = 0; ch < chunks; ++ch) {
    s += psum[(size_t)ch * Np + col];
    q += psq [(size_t)ch * Np + col];
  }
  float mu  = s / (float)NN;
  float var = q / (float)NN - mu * mu;
  mean[col] = mu;
  inv[col]  = rsqrtf(var + BN_EPS);
}

// ---------------- BN1 apply + ReLU -> bf16 (K-padded A for GEMM2) -----------
__global__ void bn1_apply(const float* __restrict__ H,
                          const float* __restrict__ mean, const float* __restrict__ inv,
                          const float* __restrict__ g, const float* __restrict__ beta,
                          bf16* __restrict__ out) {
  size_t idx = (size_t)blockIdx.x * blockDim.x + threadIdx.x;   // N * Hp
  if (idx >= (size_t)NN * Hp) return;
  int j = (int)(idx % Hp);
  float v = 0.f;
  if (j < HH) {
    v = (H[idx] - mean[j]) * inv[j] * g[j] + beta[j];
    v = fmaxf(v, 0.f);
  }
  out[idx] = (bf16)v;
}

// ---------------- BN2 apply (+optional ReLU) -> d_out slices ----------------
__global__ void bn2_apply(const float* __restrict__ H,
                          const float* __restrict__ mean, const float* __restrict__ inv,
                          const float* __restrict__ g, const float* __restrict__ beta,
                          int doRelu, float* __restrict__ xsOut, float* __restrict__ finalOut) {
  size_t idx = (size_t)blockIdx.x * blockDim.x + threadIdx.x;   // N * DD
  if (idx >= (size_t)NN * DD) return;
  int row = (int)(idx / DD);
  int j   = (int)(idx % DD);
  float v = (H[(size_t)row * Dp + j] - mean[j]) * inv[j] * g[j] + beta[j];
  if (doRelu) v = fmaxf(v, 0.f);
  xsOut[idx] = v;
  if (finalOut) finalOut[idx] = v;
}

extern "C" void kernel_launch(void* const* d_in, const int* in_sizes, int n_in,
                              void* d_out, int out_size, void* d_ws, size_t ws_size,
                              hipStream_t stream) {
  const float* x0   = (const float*)d_in[1];
  const int*   ei   = (const int*)  d_in[2];
  const float* ea   = (const float*)d_in[3];
  const float* W1   = (const float*)d_in[4];
  const float* b1   = (const float*)d_in[5];
  const float* g1   = (const float*)d_in[6];
  const float* bt1  = (const float*)d_in[7];
  const float* W2   = (const float*)d_in[8];
  const float* b2   = (const float*)d_in[9];
  const float* g2   = (const float*)d_in[10];
  const float* bt2  = (const float*)d_in[11];
  const float* epsA = (const float*)d_in[12];
  float* out = (float*)d_out;

  // ---- workspace layout (all 256B aligned) ----
  char* ws = (char*)d_ws;
  size_t off = 0;
  float* aggr = (float*)(ws + off); off += (size_t)NN * DD * 4;      // 120 MB
  bf16*  xb   = (bf16*) (ws + off); off += (size_t)NN * Dp * 2;      //  64 MB
  float* h1   = (float*)(ws + off); off += (size_t)NN * Hp * 4;      // 256 MB
  bf16*  h1b  = (bf16*) (ws + off); off += (size_t)NN * Hp * 2;      // 128 MB
  float* h2   = (float*)(ws + off); off += (size_t)NN * Dp * 4;      // 128 MB
  bf16*  Wt1  = (bf16*) (ws + off); off += (size_t)Hp * Dp * 2;
  bf16*  Wt2  = (bf16*) (ws + off); off += (size_t)Dp * Hp * 2;
  float* psum = (float*)(ws + off); off += (size_t)CHUNKS * Hp * 4;
  float* psq  = (float*)(ws + off); off += (size_t)CHUNKS * Hp * 4;
  float* mu1  = (float*)(ws + off); off += (size_t)Hp * 4;
  float* iv1  = (float*)(ws + off); off += (size_t)Hp * 4;
  float* mu2  = (float*)(ws + off); off += (size_t)Dp * 4;
  float* iv2  = (float*)(ws + off); off += (size_t)Dp * 4;

  const size_t ND = (size_t)NN * DD;
  const int TB = 256;
  const int rowsPerChunk = NN / CHUNKS;   // 1250

  for (int layer = 0; layer < LL; ++layer) {
    const float* xcur = (layer == 0) ? x0 : (out + ND + (size_t)(layer - 1) * ND);
    float* xsOut = out + ND + (size_t)layer * ND;
    float* finalOut = (layer == LL - 1) ? out : nullptr;

    // 1) aggr = segment_sum(relu(x[src] + edge_attr), dst)
    hipMemsetAsync(aggr, 0, (size_t)NN * DD * 4, stream);
    {
      size_t total = (size_t)EE * (DD / 4);
      scatter_kernel<<<(unsigned)cdiv(total, TB), TB, 0, stream>>>(xcur, ea, ei, aggr);
    }

    // 2) h0 = (1+eps)*x + aggr  -> bf16 [N,Dp]
    {
      size_t total = (size_t)NN * Dp;
      combine_kernel<<<(unsigned)cdiv(total, TB), TB, 0, stream>>>(xcur, aggr, epsA, layer, xb);
    }

    // 3) stage weights transposed to bf16 (n-major, K contiguous, zero padded)
    stage_wt<<<(unsigned)cdiv((size_t)Hp * Dp, TB), TB, 0, stream>>>(
        W1 + (size_t)layer * DD * HH, Wt1, DD, HH, Dp, Hp);
    stage_wt<<<(unsigned)cdiv((size_t)Dp * Hp, TB), TB, 0, stream>>>(
        W2 + (size_t)layer * HH * DD, Wt2, HH, DD, Hp, Dp);

    // 4) GEMM1: h1[N,Hp] = xb[N,Dp] @ W1 + b1   (bf16 WMMA, f32 accum)
    gemm_wmma<<<dim3(NN / 32, Hp / 64), 32, 0, stream>>>(
        xb, Wt1, b1 + (size_t)layer * HH, HH, h1, Dp, Hp);

    // 5) BN1 stats (deterministic two-stage) + apply + ReLU -> bf16 [N,Hp]
    bn_partial<<<dim3(Hp / 16, CHUNKS), 256, 0, stream>>>(h1, Hp, rowsPerChunk, psum, psq, Hp);
    bn_final<<<(unsigned)cdiv(Hp, 64), 64, 0, stream>>>(psum, psq, Hp, CHUNKS, mu1, iv1);
    bn1_apply<<<(unsigned)cdiv((size_t)NN * Hp, TB), TB, 0, stream>>>(
        h1, mu1, iv1, g1 + (size_t)layer * HH, bt1 + (size_t)layer * HH, h1b);

    // 6) GEMM2: h2[N,Dp] = h1b[N,Hp] @ W2 + b2
    gemm_wmma<<<dim3(NN / 32, Dp / 64), 32, 0, stream>>>(
        h1b, Wt2, b2 + (size_t)layer * DD, DD, h2, Hp, Dp);

    // 7) BN2 stats + apply (+ReLU except last layer) -> d_out slices
    bn_partial<<<dim3(Dp / 16, CHUNKS), 256, 0, stream>>>(h2, Dp, rowsPerChunk, psum, psq, Dp);
    bn_final<<<(unsigned)cdiv(Dp, 64), 64, 0, stream>>>(psum, psq, Dp, CHUNKS, mu2, iv2);
    bn2_apply<<<(unsigned)cdiv(ND, TB), TB, 0, stream>>>(
        h2, mu2, iv2, g2 + (size_t)layer * DD, bt2 + (size_t)layer * DD,
        (layer < LL - 1) ? 1 : 0, xsOut, finalOut);
  }
}
